// MixtureOfExperts_60644938220147
// MI455X (gfx1250) — matrix operand order repaired and checked by reading, not compiled, not run on hardware
//
#include <hip/hip_runtime.h>
#include <hip/hip_bf16.h>

#define BTOK 16384
#define NEXP 8
#define DIM  1024
#define ODIM 1024

#define TM 128
#define TN 64
#define TK 32
#define LDA 40   // padded LDS row stride in bf16 elems (80B -> conflict-free)
#define LDB 40

typedef __attribute__((ext_vector_type(16))) __bf16 v16bf;
typedef __attribute__((ext_vector_type(8)))  float  v8f;
typedef __attribute__((ext_vector_type(4)))  float  f32x4;
typedef __attribute__((ext_vector_type(4)))  unsigned int u32x4;

union Frag { v16bf v; u32x4 q[2]; };

// ---------------------------------------------------------------- zero pass
__global__ void moe_zero(float* __restrict__ out, int* __restrict__ counts, size_t n4) {
    size_t i = (size_t)blockIdx.x * blockDim.x + threadIdx.x;
    f32x4* o = (f32x4*)out;
    const size_t step = (size_t)gridDim.x * blockDim.x;
    for (; i < n4; i += step) o[i] = (f32x4){0.f, 0.f, 0.f, 0.f};
    if (blockIdx.x == 0 && threadIdx.x < NEXP) counts[threadIdx.x] = 0;
}

// ---------------------------------------------------------------- routing
__global__ void moe_route(const float* __restrict__ gates, int* __restrict__ counts,
                          int* __restrict__ tok, float* __restrict__ gatev) {
    int b = blockIdx.x * blockDim.x + threadIdx.x;
    if (b >= BTOK) return;
    #pragma unroll
    for (int e = 0; e < NEXP; ++e) {
        float g = gates[b * NEXP + e];
        if (g > 0.f) {
            int p = atomicAdd(&counts[e], 1);
            tok[e * BTOK + p]   = b;
            gatev[e * BTOK + p] = g;
        }
    }
}

// ---------------------------------------------------------------- expert GEMM
__global__ __launch_bounds__(256)
void moe_gemm(const float* __restrict__ x, const float* __restrict__ W,
              const float* __restrict__ bias, const int* __restrict__ counts,
              const int* __restrict__ tok, const float* __restrict__ gatev,
              float* __restrict__ out) {
    const int e     = blockIdx.z;
    const int cnt   = counts[e];
    const int mbase = blockIdx.y * TM;
    if (mbase >= cnt) return;                 // uniform: whole block exits
    const int n0    = blockIdx.x * TN;

    __shared__ __align__(16) __bf16 sAhi[TM * LDA];
    __shared__ __align__(16) __bf16 sAlo[TM * LDA];
    __shared__ __align__(16) __bf16 sBhi[TN * LDB];
    __shared__ __align__(16) __bf16 sBlo[TN * LDB];

    const int tid  = threadIdx.x;
    const int lane = tid & 31;
    const int wid  = tid >> 5;     // 0..7
    const int wm   = wid & 3;      // 4 waves over M (32 rows each)
    const int wn   = wid >> 2;     // 2 waves over N (32 cols each)
    const int half = lane >> 4;
    const int lr   = lane & 15;

    const float* We   = W + (size_t)e * DIM * ODIM;
    const int*   tokE = tok + e * BTOK;
    const float* gvE  = gatev + e * BTOK;

    // -------- hoisted staging addressing (loop-invariant) --------
    const int rpA = tid >> 3;            // A: 8 threads/row, 32 rows/pass
    const int cqA = (tid & 7) << 2;
    const int krB = tid >> 4;            // B: 16 threads/k-row, 16 k/pass
    const int nqB = (tid & 15) << 2;

    const float* xrow[4];
    bool vA[4];
    #pragma unroll
    for (int p = 0; p < 4; ++p) {
        int gr = mbase + p * 32 + rpA;
        vA[p]  = gr < cnt;
        int tk = vA[p] ? tokE[gr] : 0;   // token index loaded ONCE
        xrow[p] = x + (size_t)tk * DIM + cqA;
    }
    const float* wrow[2];
    #pragma unroll
    for (int p = 0; p < 2; ++p)
        wrow[p] = We + (size_t)(p * 16 + krB) * ODIM + n0 + nqB;

    v8f acc[2][2];
    #pragma unroll
    for (int i = 0; i < 2; ++i)
        #pragma unroll
        for (int j = 0; j < 2; ++j)
            acc[i][j] = (v8f){0.f, 0.f, 0.f, 0.f, 0.f, 0.f, 0.f, 0.f};

    // -------- software pipeline: prefetch K-tile 0 into registers --------
    f32x4 ra[4], rb[2];
    #pragma unroll
    for (int p = 0; p < 4; ++p)
        ra[p] = vA[p] ? *(const f32x4*)(xrow[p]) : (f32x4){0.f, 0.f, 0.f, 0.f};
    #pragma unroll
    for (int p = 0; p < 2; ++p)
        rb[p] = *(const f32x4*)(wrow[p]);

    for (int kk = 0; kk < DIM; kk += TK) {
        __syncthreads();
        // ---- commit registers -> LDS (f32 split to bf16 hi/lo)
        #pragma unroll
        for (int p = 0; p < 4; ++p) {
            int row = p * 32 + rpA;
            #pragma unroll
            for (int j = 0; j < 4; ++j) {
                float  f = ra[p][j];
                __bf16 h = (__bf16)f;
                __bf16 l = (__bf16)(f - (float)h);
                sAhi[row * LDA + cqA + j] = h;
                sAlo[row * LDA + cqA + j] = l;
            }
        }
        #pragma unroll
        for (int p = 0; p < 2; ++p) {
            int k = p * 16 + krB;
            #pragma unroll
            for (int j = 0; j < 4; ++j) {
                float  f = rb[p][j];
                __bf16 h = (__bf16)f;
                __bf16 l = (__bf16)(f - (float)h);
                sBhi[(nqB + j) * LDB + k] = h;   // transposed [n][k]
                sBlo[(nqB + j) * LDB + k] = l;
            }
        }
        __syncthreads();

        // ---- issue next K-tile's global loads; latency hidden under WMMAs
        if (kk + TK < DIM) {
            #pragma unroll
            for (int p = 0; p < 4; ++p)
                ra[p] = vA[p] ? *(const f32x4*)(xrow[p] + kk + TK)
                              : (f32x4){0.f, 0.f, 0.f, 0.f};
            #pragma unroll
            for (int p = 0; p < 2; ++p)
                rb[p] = *(const f32x4*)(wrow[p] + (size_t)(kk + TK) * ODIM);
            __builtin_prefetch(We + (size_t)(kk + 2 * TK) * ODIM + n0, 0, 0);
        }

        // ---- load fragments (ISA 7.12.2 16-bit layouts)
        Frag aHi[2], aLo[2], bHi[2], bLo[2];
        #pragma unroll
        for (int ti = 0; ti < 2; ++ti) {
            int row = wm * 32 + ti * 16 + lr;
            const __bf16* ph = &sAhi[row * LDA];
            const __bf16* pl = &sAlo[row * LDA];
            aHi[ti].q[0] = *(const u32x4*)(ph + 8 * half);
            aHi[ti].q[1] = *(const u32x4*)(ph + 16 + 8 * half);
            aLo[ti].q[0] = *(const u32x4*)(pl + 8 * half);
            aLo[ti].q[1] = *(const u32x4*)(pl + 16 + 8 * half);
        }
        #pragma unroll
        for (int tj = 0; tj < 2; ++tj) {
            int col = wn * 32 + tj * 16 + lr;
            const __bf16* ph = &sBhi[col * LDB];
            const __bf16* pl = &sBlo[col * LDB];
            bHi[tj].q[0] = *(const u32x4*)(ph + 8 * half);
            bHi[tj].q[1] = *(const u32x4*)(ph + 16 + 8 * half);
            bLo[tj].q[0] = *(const u32x4*)(pl + 8 * half);
            bLo[tj].q[1] = *(const u32x4*)(pl + 16 + 8 * half);
        }

        // ---- 12 WMMAs: hi*hi + hi*lo + lo*hi (bf16 split ~ fp32 fidelity)
        #pragma unroll
        for (int ti = 0; ti < 2; ++ti)
            #pragma unroll
            for (int tj = 0; tj < 2; ++tj) {
                acc[ti][tj] = __builtin_amdgcn_wmma_f32_16x16x32_bf16(
                    false, aHi[ti].v, false, bHi[tj].v, (short)0, acc[ti][tj], false, false);
                acc[ti][tj] = __builtin_amdgcn_wmma_f32_16x16x32_bf16(
                    false, aHi[ti].v, false, bLo[tj].v, (short)0, acc[ti][tj], false, false);
                acc[ti][tj] = __builtin_amdgcn_wmma_f32_16x16x32_bf16(
                    false, aLo[ti].v, false, bHi[tj].v, (short)0, acc[ti][tj], false, false);
            }
    }

    // ---- epilogue: out[token] += gate * (acc + bias); exactly 2 adds/element
    #pragma unroll
    for (int ti = 0; ti < 2; ++ti) {
        #pragma unroll
        for (int tj = 0; tj < 2; ++tj) {
            int   colg = n0 + wn * 32 + tj * 16 + lr;
            float bb   = bias[e * ODIM + colg];
            #pragma unroll
            for (int v = 0; v < 8; ++v) {
                int gr = mbase + wm * 32 + ti * 16 + v + 8 * half;
                if (gr < cnt) {
                    int   tkn = tokE[gr];
                    float g   = gvE[gr];
                    atomicAdd(out + (size_t)tkn * ODIM + colg, g * (acc[ti][tj][v] + bb));
                }
            }
        }
    }
}

// ---------------------------------------------------------------- launcher
extern "C" void kernel_launch(void* const* d_in, const int* in_sizes, int n_in,
                              void* d_out, int out_size, void* d_ws, size_t ws_size,
                              hipStream_t stream) {
    const float* x     = (const float*)d_in[0];
    const float* gates = (const float*)d_in[1];
    const float* W     = (const float*)d_in[2];
    const float* b     = (const float*)d_in[3];
    float* out = (float*)d_out;

    int*   counts = (int*)d_ws;                  // 8 (padded to 16) ints
    int*   tok    = counts + 16;                 // E*B ints
    float* gatev  = (float*)(tok + NEXP * BTOK); // E*B floats

    const size_t n4 = (size_t)BTOK * ODIM / 4;
    moe_zero<<<4096, 256, 0, stream>>>(out, counts, n4);
    moe_route<<<BTOK / 256, 256, 0, stream>>>(gates, counts, tok, gatev);

    dim3 grid(ODIM / TN, BTOK / TM, NEXP);       // (16, 128, 8); inactive M-tiles exit
    moe_gemm<<<grid, 256, 0, stream>>>(x, W, b, counts, tok, gatev, out);
}
